// Transformer_13082470383898
// MI455X (gfx1250) — matrix operand order repaired and checked by reading, not compiled, not run on hardware
//
#include <hip/hip_runtime.h>
#include <hip/hip_bf16.h>

typedef __bf16 bf16;
typedef __attribute__((ext_vector_type(16))) bf16          bf16x16;
typedef __attribute__((ext_vector_type(8)))  float         f32x8;
typedef __attribute__((ext_vector_type(4)))  unsigned int  u32x4;

// ---- model constants (match reference) ----
constexpr int VV  = 50257;
constexpr int DD  = 12;
constexpr int HH  = 16;
constexpr int HDD = 64;
constexpr int NN  = 1024;
constexpr int TT  = 512;
constexpr int BB  = 2;
constexpr int BT  = BB * TT;

union Frag16 {
  bf16x16 v;
  u32x4   q[2];
  bf16    e[16];
};

__device__ __forceinline__ float gelu_f(float x) {
  float x3 = x * x * x;
  return 0.5f * x * (1.0f + tanhf(0.79788456080286535588f * (x + 0.044715f * x3)));
}

// block-wide sum; blockDim.x <= 256, sh must hold >= 8 floats
__device__ __forceinline__ float blk_sum(float v, float* sh) {
  __syncthreads();
  #pragma unroll
  for (int o = 16; o > 0; o >>= 1) v += __shfl_down(v, o, 32);
  if ((threadIdx.x & 31) == 0) sh[threadIdx.x >> 5] = v;
  __syncthreads();
  if (threadIdx.x == 0) {
    float r = 0.f;
    int nw = (blockDim.x + 31) >> 5;
    for (int i = 0; i < nw; ++i) r += sh[i];
    sh[0] = r;
  }
  __syncthreads();
  return sh[0];
}

// ---------------- WMMA GEMM ----------------
// C[m,n] (+)= scale * ( sum_k A[m,k]*B[k,n] + bias[n] )
// A: bf16 row-major [M,K], lda.  B either:
//   BF32=true : fp32 row-major [K,N], ldb   (weights, converted in-register)
//   BF32=false: bf16 row-major Bt[N,K], ldb (Bt[n,k] = B[k,n])
// Batch over grid.z: A += z*sAz, B += z*sBz, C += (z/zdiv)*sC1 + (z%zdiv)*sC2
// Block = 32x4: 4 waves, each owning one 16-wide n-subtile.
// Each wave accumulates MT=4 m-subtiles (64 rows) against ONE shared B
// fragment per K-step, amortizing the fp32-weight gather 4x.
// Loads are branch-free (edge columns clamped); only stores are masked.
// Requires: M % 64 == 0, K % 32 == 0.
constexpr int MT = 4;

template<bool BF32, bool ACC>
__global__ void wmma_gemm(const bf16* __restrict__ A, int lda, long sAz,
                          const float* __restrict__ Bf, const bf16* __restrict__ Bb,
                          int ldb, long sBz,
                          float* __restrict__ C, int ldc, int zdiv, long sC1, long sC2,
                          int M, int Nn, int K, float scale, const float* __restrict__ bias) {
  int lane = threadIdx.x;                              // wave32 lane
  int n0 = (blockIdx.x * blockDim.y + threadIdx.y) * 16;
  int m0 = blockIdx.y * (16 * MT);
  int z  = blockIdx.z;

  int half = lane >> 4;                                // 0 or 1
  int l15  = lane & 15;
  int rowA = m0 + l15;
  int colB = n0 + l15;
  int colL = (colB < Nn) ? colB : (Nn - 1);            // clamped load column
  int koffA = half * 16;                               // A: lane<16 -> K 0..15
  int koffB = half * 16;                               // B: lane<16 -> K 0..15

  const bf16* ap = A + (long)z * sAz + (long)rowA * lda + koffA;
  const long arow16 = 16L * lda;

  f32x8 acc[MT];
  #pragma unroll
  for (int i = 0; i < MT; ++i) acc[i] = (f32x8){0.f, 0.f, 0.f, 0.f, 0.f, 0.f, 0.f, 0.f};

  if (BF32) {
    const float* bp = Bf + (long)z * sBz + (long)koffB * ldb + colL;
    const long bstep = 32L * ldb;
    for (int k = 0; k < K; k += 32) {
      Frag16 a[MT], b;
      #pragma unroll
      for (int i = 0; i < MT; ++i) {
        a[i].q[0] = *(const u32x4*)(ap + i * arow16);
        a[i].q[1] = *(const u32x4*)(ap + i * arow16 + 8);
      }
      float tmp[16];
      #pragma unroll
      for (int j = 0; j < 16; ++j) tmp[j] = bp[(long)j * ldb];
      #pragma unroll
      for (int j = 0; j < 16; ++j) b.e[j] = (bf16)tmp[j];
      #pragma unroll
      for (int i = 0; i < MT; ++i)
        acc[i] = __builtin_amdgcn_wmma_f32_16x16x32_bf16(false, a[i].v, false, b.v,
                                                         (short)0, acc[i], false, false);
      ap += 32;
      bp += bstep;
    }
  } else {
    const bf16* bp = Bb + (long)z * sBz + (long)colL * ldb + koffB;
    for (int k = 0; k < K; k += 32) {
      Frag16 a[MT], b;
      #pragma unroll
      for (int i = 0; i < MT; ++i) {
        a[i].q[0] = *(const u32x4*)(ap + i * arow16);
        a[i].q[1] = *(const u32x4*)(ap + i * arow16 + 8);
      }
      b.q[0] = *(const u32x4*)(bp);
      b.q[1] = *(const u32x4*)(bp + 8);
      #pragma unroll
      for (int i = 0; i < MT; ++i)
        acc[i] = __builtin_amdgcn_wmma_f32_16x16x32_bf16(false, a[i].v, false, b.v,
                                                         (short)0, acc[i], false, false);
      ap += 32;
      bp += 32;
    }
  }

  float bv = bias ? bias[colL] : 0.f;
  float* cp = C + (long)(z / zdiv) * sC1 + (long)(z % zdiv) * sC2;
  if (colB < Nn) {
    #pragma unroll
    for (int i = 0; i < MT; ++i) {
      #pragma unroll
      for (int r = 0; r < 8; ++r) {
        int m = m0 + i * 16 + r + half * 8;            // C: VGPR r, lanes16-31 -> M+8
        long idx = (long)m * ldc + colB;
        float val = scale * (acc[i][r] + bv);
        if (ACC) cp[idx] += val; else cp[idx] = val;
      }
    }
  }
}

// ---------------- elementwise / LN kernels ----------------
__global__ void embed_kernel(const int* __restrict__ tok, const float* __restrict__ emb,
                             const float* __restrict__ pos, float* __restrict__ x) {
  long total = (long)BT * NN;
  for (long i = (long)blockIdx.x * blockDim.x + threadIdx.x; i < total;
       i += (long)gridDim.x * blockDim.x) {
    int n = (int)(i % NN);
    long bt = i / NN;
    int t = (int)(bt % TT);
    int tk = tok[bt];
    x[i] = 1024.f * emb[(long)tk * NN + n] + 32.f * pos[(long)t * NN + n];
  }
}

// LN over row of ncols (+optional gelu) -> bf16
__global__ void ln_act_kernel(const float* __restrict__ x, const float* __restrict__ s,
                              const float* __restrict__ b, bf16* __restrict__ out,
                              int ncols, int dogelu) {
  __shared__ float sh[8];
  long row = blockIdx.x;
  const float* xr = x + row * ncols;
  float sum = 0.f, sq = 0.f;
  for (int c = threadIdx.x; c < ncols; c += blockDim.x) {
    float v = xr[c]; sum += v; sq += v * v;
  }
  sum = blk_sum(sum, sh);
  sq  = blk_sum(sq, sh);
  float mu = sum / ncols;
  float var = sq / ncols - mu * mu;
  float rs = rsqrtf(var + 1e-6f);
  for (int c = threadIdx.x; c < ncols; c += blockDim.x) {
    float y = (xr[c] - mu) * rs * s[c] + b[c];
    if (dogelu) y = gelu_f(y);
    out[row * ncols + c] = (bf16)y;
  }
}

// per-(b,t,h) LN of q and k (HD each) -> bf16 [B,H,T,HD]
__global__ void qk_ln_kernel(const float* __restrict__ qk,
                             const float* __restrict__ qs, const float* __restrict__ qb,
                             const float* __restrict__ ks, const float* __restrict__ kb,
                             bf16* __restrict__ qout, bf16* __restrict__ kout) {
  __shared__ float sh[8];
  int idx = blockIdx.x;                 // (b*T + t)*H + h
  int h = idx % HH;
  long bt = idx / HH;
  int t = (int)(bt % TT);
  int b = (int)(bt / TT);
  const float* base = qk + bt * (2L * NN) + (long)h * (2 * HDD);
  long orow = (((long)b * HH + h) * TT + t) * HDD;
  int i = threadIdx.x;                  // blockDim == 64

  float v = base[i];
  float sum = blk_sum(v, sh);
  float sq  = blk_sum(v * v, sh);
  float mu = sum / HDD, var = sq / HDD - mu * mu;
  float rs = rsqrtf(var + 1e-6f);
  qout[orow + i] = (bf16)((v - mu) * rs * qs[i] + qb[i]);

  float w = base[HDD + i];
  sum = blk_sum(w, sh);
  sq  = blk_sum(w * w, sh);
  mu = sum / HDD; var = sq / HDD - mu * mu;
  rs = rsqrtf(var + 1e-6f);
  kout[orow + i] = (bf16)((w - mu) * rs * ks[i] + kb[i]);
}

// v fp32 [B,T,H,HD] -> vT bf16 [B,H,HD,T]
__global__ void vpack_kernel(const float* __restrict__ vf, bf16* __restrict__ vT) {
  long total = (long)BT * NN;
  for (long i = (long)blockIdx.x * blockDim.x + threadIdx.x; i < total;
       i += (long)gridDim.x * blockDim.x) {
    int n = (int)(i % NN);
    long bt = i / NN;
    int t = (int)(bt % TT);
    int b = (int)(bt / TT);
    int h = n / HDD, d = n % HDD;
    vT[(((long)b * HH + h) * HDD + d) * TT + t] = (bf16)vf[i];
  }
}

// causal softmax (no max-sub, faithful): phi bf16 [B*H, T, T]; blockDim=256, T=512
__global__ void softmax_kernel(const float* __restrict__ S, bf16* __restrict__ phi) {
  __shared__ float sh[8];
  long row = blockIdx.x;                // (b*H+h)*T + q
  int qi = (int)(row % TT);
  const float* sr = S + row * (long)TT;
  bf16* pr = phi + row * (long)TT;
  int c0 = threadIdx.x, c1 = threadIdx.x + 256;
  float v0 = (c0 <= qi) ? expf(sr[c0]) : 0.f;
  float v1 = (c1 <= qi) ? expf(sr[c1]) : 0.f;
  float tot = blk_sum(v0 + v1, sh);
  float inv = 1.f / tot;
  pr[c0] = (bf16)(v0 * inv);
  pr[c1] = (bf16)(v1 * inv);
}

__global__ void f2bf_kernel(const float* __restrict__ in, bf16* __restrict__ out,
                            long n, int dogelu) {
  for (long i = (long)blockIdx.x * blockDim.x + threadIdx.x; i < n;
       i += (long)gridDim.x * blockDim.x) {
    float v = in[i];
    if (dogelu) v = gelu_f(v);
    out[i] = (bf16)v;
  }
}

// ---------------- host-side launch helpers ----------------
static void gemm_wf32(hipStream_t st, const bf16* A, int lda, long sAz,
                      const float* Bf, int ldb, long sBz,
                      float* C, int ldc, int zdiv, long sC1, long sC2,
                      int M, int Nn, int K, int Z, float scale,
                      const float* bias, bool accum) {
  dim3 grid((Nn + 63) / 64, M / (16 * MT), Z), block(32, 4);
  if (accum)
    wmma_gemm<true, true><<<grid, block, 0, st>>>(A, lda, sAz, Bf, nullptr, ldb, sBz,
                                                  C, ldc, zdiv, sC1, sC2, M, Nn, K, scale, bias);
  else
    wmma_gemm<true, false><<<grid, block, 0, st>>>(A, lda, sAz, Bf, nullptr, ldb, sBz,
                                                   C, ldc, zdiv, sC1, sC2, M, Nn, K, scale, bias);
}

static void gemm_wbf16(hipStream_t st, const bf16* A, int lda, long sAz,
                       const bf16* Bt, int ldb, long sBz,
                       float* C, int ldc, int zdiv, long sC1, long sC2,
                       int M, int Nn, int K, int Z, float scale) {
  dim3 grid((Nn + 63) / 64, M / (16 * MT), Z), block(32, 4);
  wmma_gemm<false, false><<<grid, block, 0, st>>>(A, lda, sAz, nullptr, Bt, ldb, sBz,
                                                  C, ldc, zdiv, sC1, sC2, M, Nn, K, scale, nullptr);
}

extern "C" void kernel_launch(void* const* d_in, const int* in_sizes, int n_in,
                              void* d_out, int out_size, void* d_ws, size_t ws_size,
                              hipStream_t stream) {
  (void)in_sizes; (void)n_in; (void)out_size; (void)ws_size;
  const int*   tokens = (const int*)  d_in[0];
  const float* embedW = (const float*)d_in[1];
  const float* posW   = (const float*)d_in[2];
  const float* ln1_s  = (const float*)d_in[3];
  const float* ln1_b  = (const float*)d_in[4];
  const float* qk_W   = (const float*)d_in[5];
  const float* v_W    = (const float*)d_in[6];
  const float* o_W    = (const float*)d_in[7];
  const float* qn_s   = (const float*)d_in[8];
  const float* qn_b   = (const float*)d_in[9];
  const float* kn_s   = (const float*)d_in[10];
  const float* kn_b   = (const float*)d_in[11];
  const float* ln2_s  = (const float*)d_in[12];
  const float* ln2_b  = (const float*)d_in[13];
  const float* mlp_W1 = (const float*)d_in[14];
  const float* mlp_W2 = (const float*)d_in[15];
  const float* lnf_s  = (const float*)d_in[16];
  const float* lnf_b  = (const float*)d_in[17];
  const float* out_W  = (const float*)d_in[18];
  const float* out_b  = (const float*)d_in[19];
  float* logits = (float*)d_out;

  char* ws = (char*)d_ws;
  size_t off = 0;
  auto carve = [&](size_t bytes) -> char* {
    char* p = ws + off;
    off = (off + bytes + 255) & ~(size_t)255;
    return p;
  };
  float* x    = (float*)carve((size_t)BT * NN * 4);
  bf16*  hbf  = (bf16*) carve((size_t)BT * NN * 2);
  float* qkb  = (float*)carve((size_t)BT * 2 * NN * 4);
  bf16*  qbf  = (bf16*) carve((size_t)BB * HH * TT * HDD * 2);
  bf16*  kbf  = (bf16*) carve((size_t)BB * HH * TT * HDD * 2);
  float* vf   = (float*)carve((size_t)BT * NN * 4);
  bf16*  vT   = (bf16*) carve((size_t)BB * HH * HDD * TT * 2);
  float* S    = (float*)carve((size_t)BB * HH * TT * TT * 4);
  bf16*  phi  = (bf16*) carve((size_t)BB * HH * TT * TT * 2);
  float* ocat = (float*)carve((size_t)BT * NN * 4);
  bf16*  ocbf = (bf16*) carve((size_t)BT * NN * 2);
  float* m1   = (float*)carve((size_t)BT * NN * 4);
  bf16*  m1bf = (bf16*) carve((size_t)BT * NN * 2);
  bf16*  xf   = (bf16*) carve((size_t)BT * NN * 2);

  const float inv32  = 1.0f / 32.0f;        // N^-0.5
  const float inv64  = 1.0f / 64.0f;        // HD^-1
  const float inv384 = 1.0f / 384.0f;       // (1/sqrt(N)) * (BETA/D)

  // x = 1024*embed[tok] + 32*pos
  embed_kernel<<<4096, 256, 0, stream>>>(tokens, embedW, posW, x);

  for (int l = 0; l < DD; ++l) {
    const float* qkWl = qk_W   + (size_t)l * NN * 2 * NN;
    const float* vWl  = v_W    + (size_t)l * NN * NN;
    const float* oWl  = o_W    + (size_t)l * NN * NN;
    const float* w1l  = mlp_W1 + (size_t)l * NN * NN;
    const float* w2l  = mlp_W2 + (size_t)l * NN * NN;

    // h = gelu(LN1(x)) -> bf16
    ln_act_kernel<<<BT, 256, 0, stream>>>(x, ln1_s + (size_t)l * NN, ln1_b + (size_t)l * NN,
                                          hbf, NN, 1);
    // qk = (h @ qk_W) / 32
    gemm_wf32(stream, hbf, NN, 0, qkWl, 2 * NN, 0, qkb, 2 * NN, 1, 0, 0,
              BT, 2 * NN, NN, 1, inv32, nullptr, false);
    // v = (h @ v_W) / 32
    gemm_wf32(stream, hbf, NN, 0, vWl, NN, 0, vf, NN, 1, 0, 0,
              BT, NN, NN, 1, inv32, nullptr, false);
    // per-head LN of q,k -> bf16 [B,H,T,HD]
    qk_ln_kernel<<<BT * HH, 64, 0, stream>>>(qkb,
        qn_s + (size_t)l * HDD, qn_b + (size_t)l * HDD,
        kn_s + (size_t)l * HDD, kn_b + (size_t)l * HDD, qbf, kbf);
    // v -> vT bf16 [B,H,HD,T]
    vpack_kernel<<<4096, 256, 0, stream>>>(vf, vT);
    // S = q @ k^T / 64, batched over B*H
    gemm_wbf16(stream, qbf, HDD, (long)TT * HDD, kbf, HDD, (long)TT * HDD,
               S, TT, 1, (long)TT * TT, 0, TT, TT, HDD, BB * HH, inv64);
    // phi = causal-softmax(S) -> bf16
    softmax_kernel<<<BB * HH * TT, 256, 0, stream>>>(S, phi);
    // ocat[b,t, h*HD+d] = sum_k phi[z,t,k] * vT[z,d,k]
    gemm_wbf16(stream, phi, TT, (long)TT * TT, vT, TT, (long)HDD * TT,
               ocat, NN, HH, (long)TT * NN, HDD, TT, HDD, TT, BB * HH, 1.0f);
    // ocat -> bf16
    f2bf_kernel<<<4096, 256, 0, stream>>>(ocat, ocbf, (long)BT * NN, 0);
    // x += (ocat @ o_W) / 384
    gemm_wf32(stream, ocbf, NN, 0, oWl, NN, 0, x, NN, 1, 0, 0,
              BT, NN, NN, 1, inv384, nullptr, true);
    // h2 = gelu(LN2(x)) -> bf16
    ln_act_kernel<<<BT, 256, 0, stream>>>(x, ln2_s + (size_t)l * NN, ln2_b + (size_t)l * NN,
                                          hbf, NN, 1);
    // m1 = (h2 @ W1) / 32
    gemm_wf32(stream, hbf, NN, 0, w1l, NN, 0, m1, NN, 1, 0, 0,
              BT, NN, NN, 1, inv32, nullptr, false);
    // m1 -> gelu -> bf16
    f2bf_kernel<<<4096, 256, 0, stream>>>(m1, m1bf, (long)BT * NN, 1);
    // x += (m1 @ W2) / 384
    gemm_wf32(stream, m1bf, NN, 0, w2l, NN, 0, x, NN, 1, 0, 0,
              BT, NN, NN, 1, inv384, nullptr, true);
  }

  // xf = LN_f(x) -> bf16
  ln_act_kernel<<<BT, 256, 0, stream>>>(x, lnf_s, lnf_b, xf, NN, 0);
  // logits = (xf @ out_W + out_b) / 32
  gemm_wf32(stream, xf, NN, 0, out_W, VV, 0, logits, VV, 1, 0, 0,
            BT, VV, NN, 1, inv32, out_b, false);
}
